// EmotionModel_20787641712805
// MI455X (gfx1250) — matrix operand order repaired and checked by reading, not compiled, not run on hardware
//
#include <hip/hip_runtime.h>
#include <stdint.h>
#include <stddef.h>

// ---------------- constants (match reference) ----------------
#define BF_      2048          // B * F_
#define CIN      256
#define SRC      1024
#define KCB      64
#define CTX      32
#define NHEAD    8
#define DHEAD    128
#define N3       (CTX * SRC)   // 32768
#define XROWS    (BF_ * CTX)   // 65536

typedef __attribute__((ext_vector_type(16))) __bf16 v16bf;
typedef __attribute__((ext_vector_type(8)))  __bf16 v8bf;
typedef __attribute__((ext_vector_type(8)))  float  v8f;

static __device__ __forceinline__ __bf16 to_bf16(float f) {
    unsigned int u = __builtin_bit_cast(unsigned int, f);
    unsigned int r = u + 0x7FFFu + ((u >> 16) & 1u);
    unsigned short h = (unsigned short)(r >> 16);
    return __builtin_bit_cast(__bf16, h);
}

// ---------------- fp32 -> bf16 convert ----------------
__global__ __launch_bounds__(256) void f2b_kernel(const float* __restrict__ s,
                                                  __bf16* __restrict__ d, size_t n) {
    size_t i = (size_t)blockIdx.x * blockDim.x + threadIdx.x;
    size_t stride = (size_t)gridDim.x * blockDim.x;
    for (; i < n; i += stride) d[i] = to_bf16(s[i]);
}

__global__ void loss_init_kernel(float* loss) { *loss = 0.0f; }

// ---------------- WMMA GEMM: C(M,N) = A(M,K) * W(N,K)^T + bias ----------------
// workgroup tile 128x128 (8 waves, 4x2), wave tile 32x64 (2x4 acc of 16x16)
// OUT_F32: write fp32, else write bf16. Bias always applied.
template <bool RELU, bool OUT_F32>
__global__ __launch_bounds__(256) void gemm_bf16_wmma(
    const __bf16* __restrict__ A, const __bf16* __restrict__ W,
    const float* __restrict__ bias, void* __restrict__ outP,
    int M, int N, int K)
{
    const int lane = threadIdx.x & 31;
    const int wave = threadIdx.x >> 5;
    const int wm   = wave >> 1;            // 0..3
    const int wn   = wave & 1;             // 0..1
    const int half = (lane >> 4) & 1;
    const int l16  = lane & 15;

    const int m_base = blockIdx.y * 128 + wm * 32;
    const int n_base = blockIdx.x * 128 + wn * 64;

    const v8f vzero = {0.f,0.f,0.f,0.f,0.f,0.f,0.f,0.f};
    v8f acc[2][4];
    #pragma unroll
    for (int i = 0; i < 2; ++i)
        #pragma unroll
        for (int j = 0; j < 4; ++j) acc[i][j] = vzero;

    for (int k0 = 0; k0 < K; k0 += 32) {
        v16bf a[2], b[4];
        #pragma unroll
        for (int tm = 0; tm < 2; ++tm) {
            const __bf16* ap = A + (size_t)(m_base + tm * 16 + l16) * K + k0 + half * 8;
            v8bf lo = *(const v8bf*)ap;
            v8bf hi = *(const v8bf*)(ap + 16);
            #pragma unroll
            for (int j = 0; j < 8; ++j) { a[tm][j] = lo[j]; a[tm][8 + j] = hi[j]; }
            __builtin_prefetch(ap + 64, 0, 1);            // next K-block of A
        }
        #pragma unroll
        for (int tn = 0; tn < 4; ++tn) {
            const __bf16* bp = W + (size_t)(n_base + tn * 16 + l16) * K + k0 + half * 8;
            v8bf blo = *(const v8bf*)bp;
            v8bf bhi = *(const v8bf*)(bp + 16);
            #pragma unroll
            for (int j = 0; j < 8; ++j) { b[tn][j] = blo[j]; b[tn][8 + j] = bhi[j]; }
            __builtin_prefetch(bp + 64, 0, 1);            // next K-block of W stream
        }
        #pragma unroll
        for (int tm = 0; tm < 2; ++tm)
            #pragma unroll
            for (int tn = 0; tn < 4; ++tn)
                acc[tm][tn] = __builtin_amdgcn_wmma_f32_16x16x32_bf16(
                    false, a[tm], false, b[tn], (short)0, acc[tm][tn], false, false);
    }

    #pragma unroll
    for (int tm = 0; tm < 2; ++tm)
        #pragma unroll
        for (int tn = 0; tn < 4; ++tn) {
            const int n = n_base + tn * 16 + l16;
            const float bv = bias[n];
            #pragma unroll
            for (int r = 0; r < 8; ++r) {
                const int m = m_base + tm * 16 + half * 8 + r;
                float v = acc[tm][tn][r] + bv;
                if (RELU) v = fmaxf(v, 0.0f);
                const size_t idx = (size_t)m * N + n;
                if (OUT_F32) ((float*)outP)[idx] = v;
                else         ((__bf16*)outP)[idx] = to_bf16(v);
            }
        }
}

// ---------------- VQ: dists, argmin, gather -> bf16, loss ----------------
__global__ __launch_bounds__(256) void vq_kernel(
    const float* __restrict__ z, const float* __restrict__ cb,
    __bf16* __restrict__ quantb, float* __restrict__ loss)
{
    __shared__ float zs[SRC];
    __shared__ float red[256];
    __shared__ float sc[KCB];
    __shared__ int   sidx;
    const int row = blockIdx.x;
    const float* zr = z + (size_t)row * SRC;
    #pragma unroll
    for (int i = 0; i < 4; ++i) zs[threadIdx.x + i * 256] = zr[threadIdx.x + i * 256];
    __syncthreads();

    const int c = threadIdx.x >> 2, p = threadIdx.x & 3;   // 4 threads / code
    const float* cr = cb + (size_t)c * SRC + p * 256;
    const float* zp = zs + p * 256;
    float dot = 0.f, cn = 0.f;
    for (int i = 0; i < 256; ++i) { float w = cr[i]; dot += w * zp[i]; cn += w * w; }
    red[threadIdx.x] = cn - 2.0f * dot;                    // ||c||^2 - 2 z.c (||z||^2 const)
    __syncthreads();
    if (p == 0) sc[c] = red[threadIdx.x] + red[threadIdx.x + 1] +
                        red[threadIdx.x + 2] + red[threadIdx.x + 3];
    __syncthreads();
    if (threadIdx.x == 0) {
        float best = sc[0]; int bi = 0;
        for (int i = 1; i < KCB; ++i) if (sc[i] < best) { best = sc[i]; bi = i; }
        sidx = bi;
    }
    __syncthreads();
    const float* q = cb + (size_t)sidx * SRC;
    float local = 0.f;
    #pragma unroll
    for (int i = 0; i < 4; ++i) {
        const int e = threadIdx.x + i * 256;
        const float qq = q[e];
        quantb[(size_t)row * SRC + e] = to_bf16(qq);
        const float d = qq - zs[e];
        local += d * d;
    }
    red[threadIdx.x] = local; __syncthreads();
    for (int o = 128; o > 0; o >>= 1) {
        if (threadIdx.x < o) red[threadIdx.x] += red[threadIdx.x + o];
        __syncthreads();
    }
    // vq_loss = q_loss + 0.25*e_loss = 1.25 * mean((quant-z)^2)
    if (threadIdx.x == 0)
        atomicAdd(loss, red[0] * (1.25f / ((float)BF_ * (float)SRC)));
}

// ---------------- LayerNorm rows of 1024, fp32 in -> bf16 out ----------------
__global__ __launch_bounds__(256) void ln_kernel(
    const float* __restrict__ x, const float* __restrict__ g,
    const float* __restrict__ b, __bf16* __restrict__ out)
{
    __shared__ float red[256];
    const int row = blockIdx.x;
    const float* xr = x + (size_t)row * SRC;
    float v[4], s = 0.f, ss = 0.f;
    #pragma unroll
    for (int i = 0; i < 4; ++i) {
        v[i] = xr[threadIdx.x + i * 256];
        s += v[i]; ss += v[i] * v[i];
    }
    red[threadIdx.x] = s; __syncthreads();
    for (int o = 128; o > 0; o >>= 1) {
        if (threadIdx.x < o) red[threadIdx.x] += red[threadIdx.x + o];
        __syncthreads();
    }
    const float mean = red[0] * (1.0f / SRC);
    __syncthreads();
    red[threadIdx.x] = ss; __syncthreads();
    for (int o = 128; o > 0; o >>= 1) {
        if (threadIdx.x < o) red[threadIdx.x] += red[threadIdx.x + o];
        __syncthreads();
    }
    const float var = red[0] * (1.0f / SRC) - mean * mean;
    const float rs = rsqrtf(var + 1e-5f);
    #pragma unroll
    for (int i = 0; i < 4; ++i) {
        const int col = threadIdx.x + i * 256;
        out[(size_t)row * SRC + col] = to_bf16((v[i] - mean) * rs * g[col] + b[col]);
    }
}

// ---------------- attention: one (frame, head) per 128-thread block ----------------
__global__ __launch_bounds__(128) void attn_kernel(
    const __bf16* __restrict__ Qb, const __bf16* __restrict__ Kb,
    const __bf16* __restrict__ Vb, float* __restrict__ out)
{
    __shared__ float  S[CTX][CTX + 1];
    __shared__ __bf16 P[CTX][CTX + 2];
    __shared__ __bf16 Vt[CTX][DHEAD + 4];
    const int bf   = blockIdx.x >> 3;
    const int h    = blockIdx.x & 7;
    const int lane = threadIdx.x & 31;
    const int wave = threadIdx.x >> 5;
    const int half = lane >> 4, l16 = lane & 15;
    const v8f vzero = {0.f,0.f,0.f,0.f,0.f,0.f,0.f,0.f};

    // stage V tile (32 x 128) into LDS, coalesced
    for (int k = 0; k < CTX; ++k)
        Vt[k][threadIdx.x] = Vb[(size_t)(bf * CTX + k) * SRC + h * DHEAD + threadIdx.x];

    // ---- scores S = Q K^T / sqrt(DH): 4 waves -> 2x2 tiles of 16x16, K=128 ----
    int tm = wave >> 1, tn = wave & 1;
    v8f acc = vzero;
    #pragma unroll
    for (int ks = 0; ks < DHEAD; ks += 32) {
        const __bf16* qp = Qb + (size_t)(bf * CTX + tm * 16 + l16) * SRC + h * DHEAD + ks + half * 8;
        v8bf lo = *(const v8bf*)qp;
        v8bf hi = *(const v8bf*)(qp + 16);
        v16bf a;
        #pragma unroll
        for (int j = 0; j < 8; ++j) { a[j] = lo[j]; a[8 + j] = hi[j]; }
        const __bf16* kp = Kb + (size_t)(bf * CTX + tn * 16 + l16) * SRC + h * DHEAD + ks + half * 8;
        v8bf klo = *(const v8bf*)kp;
        v8bf khi = *(const v8bf*)(kp + 16);
        v16bf b;
        #pragma unroll
        for (int j = 0; j < 8; ++j) { b[j] = klo[j]; b[8 + j] = khi[j]; }
        acc = __builtin_amdgcn_wmma_f32_16x16x32_bf16(false, a, false, b, (short)0, acc, false, false);
    }
    #pragma unroll
    for (int r = 0; r < 8; ++r)
        S[tm * 16 + half * 8 + r][tn * 16 + l16] = acc[r] * 0.0883883476483184f; // 1/sqrt(128)
    __syncthreads();

    // ---- softmax over the 32 keys, one thread per query row ----
    if (threadIdx.x < CTX) {
        const int rr = threadIdx.x;
        float mx = S[rr][0];
        #pragma unroll
        for (int j = 1; j < CTX; ++j) mx = fmaxf(mx, S[rr][j]);
        float e[CTX], sum = 0.f;
        #pragma unroll
        for (int j = 0; j < CTX; ++j) { e[j] = __expf(S[rr][j] - mx); sum += e[j]; }
        const float inv = 1.0f / sum;
        #pragma unroll
        for (int j = 0; j < CTX; ++j) P[rr][j] = to_bf16(e[j] * inv);
    }
    __syncthreads();

    // ---- O = P(32x32) @ Vhead(32x128): 16 tiles, 4 per wave, K=32 (one WMMA each) ----
    const int ng = wave & 1;
    tm = wave >> 1;
    v16bf a;
    #pragma unroll
    for (int j = 0; j < 16; ++j) {
        const int kk = half * 8 + ((j < 8) ? 0 : 16) + (j & 7);
        a[j] = P[tm * 16 + l16][kk];
    }
    #pragma unroll
    for (int i = 0; i < 4; ++i) {
        const int tn2 = ng * 4 + i;
        v16bf b;
        #pragma unroll
        for (int j = 0; j < 16; ++j) b[j] = Vt[half * 8 + ((j < 8) ? 0 : 16) + (j & 7)][tn2 * 16 + l16];
        v8f o = __builtin_amdgcn_wmma_f32_16x16x32_bf16(false, a, false, b, (short)0, vzero, false, false);
        #pragma unroll
        for (int r = 0; r < 8; ++r) {
            const int m = tm * 16 + half * 8 + r;
            out[(size_t)(bf * CTX + m) * SRC + h * DHEAD + tn2 * 16 + l16] = o[r];
        }
    }
}

// ================== host launcher ==================
extern "C" void kernel_launch(void* const* d_in, const int* in_sizes, int n_in,
                              void* d_out, int out_size, void* d_ws, size_t ws_size,
                              hipStream_t stream) {
    const float* emo_prompts = (const float*)d_in[0];
    const float* fc_w  = (const float*)d_in[1];
    const float* fc_b  = (const float*)d_in[2];
    const float* codebook = (const float*)d_in[3];
    const float* emo_w1 = (const float*)d_in[4];  const float* emo_b1 = (const float*)d_in[5];
    const float* emo_w2 = (const float*)d_in[6];  const float* emo_b2 = (const float*)d_in[7];
    const float* emo_w3 = (const float*)d_in[8];  const float* emo_b3 = (const float*)d_in[9];
    const float* emo_ln_g = (const float*)d_in[10]; const float* emo_ln_b = (const float*)d_in[11];
    const float* kv_w1 = (const float*)d_in[12];  const float* kv_b1 = (const float*)d_in[13];
    const float* kv_w2 = (const float*)d_in[14];  const float* kv_b2 = (const float*)d_in[15];
    const float* kv_w3 = (const float*)d_in[16];  const float* kv_b3 = (const float*)d_in[17];
    const float* kv_ln_g = (const float*)d_in[18]; const float* kv_ln_b = (const float*)d_in[19];
    const float* q_w = (const float*)d_in[20];    const float* q_b = (const float*)d_in[21];
    const float* k_w = (const float*)d_in[22];    const float* k_b = (const float*)d_in[23];
    const float* v_w = (const float*)d_in[24];    const float* v_b = (const float*)d_in[25];

    // ---- workspace layout (bytes) ----
    char* ws = (char*)d_ws;
    size_t off = 0;
    auto take = [&](size_t bytes) { char* p = ws + off; off += bytes; return p; };
    __bf16* wb_fc = (__bf16*)take((size_t)SRC * CIN * 2);
    __bf16* wb_e1 = (__bf16*)take((size_t)1024 * CIN * 2);
    __bf16* wb_e2 = (__bf16*)take((size_t)1024 * 1024 * 2);
    __bf16* wb_e3 = (__bf16*)take((size_t)N3 * 1024 * 2);
    __bf16* wb_k1 = (__bf16*)take((size_t)1024 * SRC * 2);
    __bf16* wb_k2 = (__bf16*)take((size_t)1024 * 1024 * 2);
    __bf16* wb_k3 = (__bf16*)take((size_t)N3 * 1024 * 2);
    __bf16* wb_q  = (__bf16*)take((size_t)SRC * SRC * 2);
    __bf16* wb_kw = (__bf16*)take((size_t)SRC * SRC * 2);
    __bf16* wb_v  = (__bf16*)take((size_t)SRC * SRC * 2);
    __bf16* xb    = (__bf16*)take((size_t)BF_ * CIN * 2);
    float*  zbuf  = (float*) take((size_t)BF_ * SRC * 4);
    __bf16* quantb= (__bf16*)take((size_t)BF_ * SRC * 2);
    __bf16* h1b   = (__bf16*)take((size_t)BF_ * 1024 * 2);
    __bf16* h2b   = (__bf16*)take((size_t)BF_ * 1024 * 2);
    float*  h3    = (float*) take((size_t)BF_ * N3 * 4);
    __bf16* xqb   = (__bf16*)take((size_t)XROWS * SRC * 2);
    __bf16* kvb   = (__bf16*)take((size_t)XROWS * SRC * 2);
    __bf16* Qbuf  = (__bf16*)take((size_t)XROWS * SRC * 2);
    __bf16* Kbuf  = (__bf16*)take((size_t)XROWS * SRC * 2);
    __bf16* Vbuf  = (__bf16*)take((size_t)XROWS * SRC * 2);
    (void)ws_size; (void)n_in; (void)in_sizes;

    float* out_f  = (float*)d_out;
    float* loss_p = out_f + (out_size - 1);

    auto f2b = [&](const float* s, __bf16* d, size_t n) {
        int grid = (int)((n + 1023) / 1024); if (grid > 8192) grid = 8192;
        f2b_kernel<<<grid, 256, 0, stream>>>(s, d, n);
    };

    // weight / input conversions
    f2b(emo_prompts, xb, (size_t)BF_ * CIN);
    f2b(fc_w,  wb_fc, (size_t)SRC * CIN);
    f2b(emo_w1, wb_e1, (size_t)1024 * CIN);
    f2b(emo_w2, wb_e2, (size_t)1024 * 1024);
    f2b(emo_w3, wb_e3, (size_t)N3 * 1024);
    f2b(kv_w1, wb_k1, (size_t)1024 * SRC);
    f2b(kv_w2, wb_k2, (size_t)1024 * 1024);
    f2b(kv_w3, wb_k3, (size_t)N3 * 1024);
    f2b(q_w, wb_q,  (size_t)SRC * SRC);
    f2b(k_w, wb_kw, (size_t)SRC * SRC);
    f2b(v_w, wb_v,  (size_t)SRC * SRC);

    // z = prompts @ fc_w^T + fc_b  (fp32 out for VQ)
    gemm_bf16_wmma<false, true><<<dim3(SRC/128, BF_/128), 256, 0, stream>>>(
        xb, wb_fc, fc_b, zbuf, BF_, SRC, CIN);

    // VQ + loss
    loss_init_kernel<<<1, 1, 0, stream>>>(loss_p);
    vq_kernel<<<BF_, 256, 0, stream>>>(zbuf, codebook, quantb, loss_p);

    // xq branch MLP
    gemm_bf16_wmma<true, false><<<dim3(1024/128, BF_/128), 256, 0, stream>>>(
        xb, wb_e1, emo_b1, h1b, BF_, 1024, CIN);
    gemm_bf16_wmma<true, false><<<dim3(1024/128, BF_/128), 256, 0, stream>>>(
        h1b, wb_e2, emo_b2, h2b, BF_, 1024, 1024);
    gemm_bf16_wmma<false, true><<<dim3(N3/128, BF_/128), 256, 0, stream>>>(
        h2b, wb_e3, emo_b3, h3, BF_, N3, 1024);
    ln_kernel<<<XROWS, 256, 0, stream>>>(h3, emo_ln_g, emo_ln_b, xqb);

    // kv branch MLP (input = straight-through quantized = codebook gather)
    gemm_bf16_wmma<true, false><<<dim3(1024/128, BF_/128), 256, 0, stream>>>(
        quantb, wb_k1, kv_b1, h1b, BF_, 1024, SRC);
    gemm_bf16_wmma<true, false><<<dim3(1024/128, BF_/128), 256, 0, stream>>>(
        h1b, wb_k2, kv_b2, h2b, BF_, 1024, 1024);
    gemm_bf16_wmma<false, true><<<dim3(N3/128, BF_/128), 256, 0, stream>>>(
        h2b, wb_k3, kv_b3, h3, BF_, N3, 1024);
    ln_kernel<<<XROWS, 256, 0, stream>>>(h3, kv_ln_g, kv_ln_b, kvb);

    // Q/K/V projections
    gemm_bf16_wmma<false, false><<<dim3(SRC/128, XROWS/128), 256, 0, stream>>>(
        xqb, wb_q, q_b, Qbuf, XROWS, SRC, SRC);
    gemm_bf16_wmma<false, false><<<dim3(SRC/128, XROWS/128), 256, 0, stream>>>(
        kvb, wb_kw, k_b, Kbuf, XROWS, SRC, SRC);
    gemm_bf16_wmma<false, false><<<dim3(SRC/128, XROWS/128), 256, 0, stream>>>(
        kvb, wb_v, v_b, Vbuf, XROWS, SRC, SRC);

    // attention -> d_out
    attn_kernel<<<BF_ * NHEAD, 128, 0, stream>>>(Qbuf, Kbuf, Vbuf, out_f);
}
